// GNN_node_18433999635114
// MI455X (gfx1250) — matrix operand order repaired and checked by reading, not compiled, not run on hardware
//
#include <hip/hip_runtime.h>
#include <hip/hip_bf16.h>

typedef __bf16 bf16;
typedef __attribute__((ext_vector_type(4)))  bf16  v4bf;
typedef __attribute__((ext_vector_type(8)))  bf16  v8bf;
typedef __attribute__((ext_vector_type(16))) bf16  v16bf;
typedef __attribute__((ext_vector_type(8)))  float v8f;

__device__ __forceinline__ void atomAddF(float* p, float v) {
    // hardware global_atomic_add_f32 path
    unsafeAtomicAdd(p, v);
}

// ---------------- utility kernels ----------------

__global__ void zero_f32(float* __restrict__ p, long n) {
    long i = (long)blockIdx.x * blockDim.x + threadIdx.x;
    if (i < n) p[i] = 0.0f;
}

__global__ void conv_bf16(const float* __restrict__ s, bf16* __restrict__ d, long n) {
    long i = (long)blockIdx.x * blockDim.x + threadIdx.x;
    if (i < n) d[i] = (bf16)s[i];
}

__global__ void init_h(const int* __restrict__ x, const float* __restrict__ emb,
                       float* __restrict__ h, long n) {
    long i = (long)blockIdx.x * blockDim.x + threadIdx.x;
    if (i >= n) return;
    int node = (int)(i >> 7);
    int d    = (int)(i & 127);
    h[i] = emb[(long)x[node] * 128 + d];
}

// out = (1+eps)*h + agg, stored as bf16 A-matrix for GEMM1
__global__ void node_pre(const float* __restrict__ h, const float* __restrict__ agg,
                         const float* __restrict__ epsArr, int l,
                         bf16* __restrict__ A1, long n) {
    long i = (long)blockIdx.x * blockDim.x + threadIdx.x;
    if (i >= n) return;
    float c = 1.0f + epsArr[l];
    A1[i] = (bf16)(c * h[i] + agg[i]);
}

// h = z*scale[c] + bias[c] (folded BatchNorm), optional relu
__global__ void bn_apply(const float* __restrict__ z, const float* __restrict__ scale,
                         const float* __restrict__ bias, float* __restrict__ dst,
                         long n, int relu) {
    long i = (long)blockIdx.x * blockDim.x + threadIdx.x;
    if (i >= n) return;
    int c = (int)(i & 127);
    float v = z[i] * scale[c] + bias[c];
    if (relu) v = fmaxf(v, 0.0f);
    dst[i] = v;
}

// ---------------- fused edge kernel ----------------
// one wave per edge; lane handles 4 of 128 dims.
// ee = edge_attr @ We + be (K=7, We in LDS), msg = relu(h[src]+ee), atomic scatter to agg[dst].
__global__ __launch_bounds__(256) void edge_gin(
    const int* __restrict__ ei, const float* __restrict__ ea,
    const float* __restrict__ We_l, const float* __restrict__ be_l,
    const float* __restrict__ h, float* __restrict__ agg, int E) {
    __shared__ float sWe[7 * 128];
    __shared__ float sbe[128];
    for (int i = threadIdx.x; i < 7 * 128 + 128; i += blockDim.x) {
        if (i < 7 * 128) sWe[i] = We_l[i];
        else             sbe[i - 7 * 128] = be_l[i - 7 * 128];
    }
    __syncthreads();
    int gt   = blockIdx.x * blockDim.x + threadIdx.x;
    int e    = gt >> 5;
    int lane = threadIdx.x & 31;
    if (e >= E) return;
    float av = (lane < 7) ? ea[(long)e * 7 + lane] : 0.0f;
    float a0 = __shfl(av, 0, 32), a1 = __shfl(av, 1, 32), a2 = __shfl(av, 2, 32),
          a3 = __shfl(av, 3, 32), a4 = __shfl(av, 4, 32), a5 = __shfl(av, 5, 32),
          a6 = __shfl(av, 6, 32);
    int src = ei[e], dst = ei[E + e];
    int d0  = lane * 4;
    const float4 hv = *(const float4*)(h + (long)src * 128 + d0);
    float hx[4] = {hv.x, hv.y, hv.z, hv.w};
    float* aggp = agg + (long)dst * 128 + d0;
#pragma unroll
    for (int j = 0; j < 4; ++j) {
        int d = d0 + j;
        float ee = sbe[d] + a0 * sWe[0 * 128 + d] + a1 * sWe[1 * 128 + d]
                 + a2 * sWe[2 * 128 + d] + a3 * sWe[3 * 128 + d]
                 + a4 * sWe[4 * 128 + d] + a5 * sWe[5 * 128 + d]
                 + a6 * sWe[6 * 128 + d];
        float m = hx[j] + ee;
        m = m > 0.0f ? m : 0.0f;
        atomAddF(aggp + j, m);
    }
}

// ---------------- WMMA GEMM ----------------
// C[Nrows,NOUT] = A[Nrows,K] * W[K,NOUT] + bias
// FUSE: A built on the fly from f32 z via relu(z*bnS+bnB) -> bf16 during LDS staging.
// block: 128 threads (4 waves); block tile 64x64; K staged in 128-wide chunks.
template <int K, int NOUT, bool FUSE>
__global__ __launch_bounds__(128) void gemm_wmma(
    const bf16* __restrict__ Abf, const float* __restrict__ Af32,
    const float* __restrict__ bnS, const float* __restrict__ bnB,
    const bf16* __restrict__ W, const float* __restrict__ bias,
    float* __restrict__ Cout, int Nrows) {
    constexpr int KT = 128;
    __shared__ alignas(32) bf16 As[64 * (KT + 8)];   // +8 bf16 pad kills bank conflicts
    __shared__ alignas(32) bf16 Bs[KT * 64];         // fragment-packed layout
    const int tid = threadIdx.x;
    const int rowBase = blockIdx.x * 64;
    const int colBase = blockIdx.y * 64;
    const int w    = tid >> 5;
    const int lane = tid & 31;
    const int lm   = lane & 15;
    const int khc  = lane >> 4;
    v8f acc[4] = {};

    for (int kt = 0; kt < K; kt += KT) {
        // stage A (64 x 128 bf16), vectorized by 4
        for (int i = tid; i < 64 * (KT / 4); i += 128) {
            int r  = i / (KT / 4);
            int c4 = i % (KT / 4);
            int row = rowBase + r;
            int cg  = kt + c4 * 4;
            v4bf o;
            if (FUSE) {
                float4 v = make_float4(0.f, 0.f, 0.f, 0.f);
                if (row < Nrows) v = *(const float4*)(Af32 + (long)row * K + cg);
                float4 sv = *(const float4*)(bnS + cg);
                float4 bv = *(const float4*)(bnB + cg);
                o[0] = (bf16)fmaxf(v.x * sv.x + bv.x, 0.f);
                o[1] = (bf16)fmaxf(v.y * sv.y + bv.y, 0.f);
                o[2] = (bf16)fmaxf(v.z * sv.z + bv.z, 0.f);
                o[3] = (bf16)fmaxf(v.w * sv.w + bv.w, 0.f);
            } else {
                if (row < Nrows) o = *(const v4bf*)(Abf + (long)row * K + cg);
                else { o[0] = o[1] = o[2] = o[3] = (bf16)0.f; }
            }
            *(v4bf*)&As[r * (KT + 8) + c4 * 4] = o;
        }
        // stage B into fragment-friendly layout [kb][n][kh][t]
        for (int i = tid; i < KT * 64; i += 128) {
            int k = i / 64, n = i % 64;
            int kb = k >> 5, kh2 = (k >> 3) & 1, t = (k & 7) | ((k >> 1) & 8);
            Bs[((kb * 64 + n) * 2 + kh2) * 16 + t] = W[(long)(kt + k) * NOUT + colBase + n];
        }
        __syncthreads();

        const int arow = w * 16 + lm;
#pragma unroll
        for (int kb = 0; kb < KT / 32; ++kb) {
            const bf16* ap = &As[arow * (KT + 8) + kb * 32 + khc * 8];
            v8bf alo = *(const v8bf*)ap;
            v8bf ahi = *(const v8bf*)(ap + 16);
            v16bf af = __builtin_shufflevector(alo, ahi,
                0, 1, 2, 3, 4, 5, 6, 7, 8, 9, 10, 11, 12, 13, 14, 15);
#pragma unroll
            for (int nt = 0; nt < 4; ++nt) {
                v16bf bf = *(const v16bf*)&Bs[((kb * 64 + nt * 16 + lm) * 2 + khc) * 16];
                acc[nt] = __builtin_amdgcn_wmma_f32_16x16x32_bf16(
                    false, af, false, bf, (short)0, acc[nt], false, false);
            }
        }
        __syncthreads();
    }

    // C/D layout: VGPR r, lanes 0-15 -> M=r ; lanes 16-31 -> M=r+8
    const int rOut = rowBase + w * 16 + khc * 8;
#pragma unroll
    for (int nt = 0; nt < 4; ++nt) {
        int col = colBase + nt * 16 + lm;
        float bcol = bias[col];
#pragma unroll
        for (int r = 0; r < 8; ++r) {
            int row = rOut + r;
            if (row < Nrows) Cout[(long)row * NOUT + col] = acc[nt][r] + bcol;
        }
    }
}

// ---------------- BatchNorm column statistics ----------------
template <int C>
__global__ void col_stats(const float* __restrict__ z, float* __restrict__ sum,
                          float* __restrict__ sq, int Nrows) {
    int col = threadIdx.x;           // blockDim.x == C
    int r0 = blockIdx.x * 128;
    int r1 = min(r0 + 128, Nrows);
    float s = 0.f, q = 0.f;
    for (int r = r0; r < r1; ++r) {
        float v = z[(long)r * C + col];
        s += v;
        q += v * v;
    }
    atomAddF(&sum[col], s);
    atomAddF(&sq[col], q);
}

template <int C>
__global__ void bn_finalize(const float* __restrict__ sum, const float* __restrict__ sq,
                            const float* __restrict__ g, const float* __restrict__ bt,
                            float* __restrict__ scale, float* __restrict__ bias, int Nrows) {
    int c = threadIdx.x;
    float inv_n = 1.0f / (float)Nrows;
    float mu  = sum[c] * inv_n;
    float var = sq[c] * inv_n - mu * mu;
    float s   = g[c] * rsqrtf(var + 1e-5f);
    scale[c] = s;
    bias[c]  = bt[c] - mu * s;
}

// ---------------- host side ----------------

extern "C" void kernel_launch(void* const* d_in, const int* in_sizes, int n_in,
                              void* d_out, int out_size, void* d_ws, size_t ws_size,
                              hipStream_t stream) {
    const int*   x   = (const int*)d_in[0];
    const int*   ei  = (const int*)d_in[1];
    const float* ea  = (const float*)d_in[2];
    const float* emb = (const float*)d_in[3];
    const float* We  = (const float*)d_in[4];
    const float* be  = (const float*)d_in[5];
    const float* eps = (const float*)d_in[6];
    const float* W1  = (const float*)d_in[7];
    const float* b1  = (const float*)d_in[8];
    const float* g1  = (const float*)d_in[9];
    const float* bt1 = (const float*)d_in[10];
    const float* W2  = (const float*)d_in[11];
    const float* b2  = (const float*)d_in[12];
    const float* gbn = (const float*)d_in[13];
    const float* bbn = (const float*)d_in[14];
    float* out = (float*)d_out;

    const int N  = in_sizes[0];
    const int E  = in_sizes[1] / 2;
    const int L  = in_sizes[6];
    const int D  = 128, D2 = 256;
    const long nND = (long)N * D;

    char* w = (char*)d_ws;
    auto alloc = [&](size_t bytes) {
        char* p = w;
        w += (bytes + 255) & ~(size_t)255;
        return p;
    };
    float* h     = (float*)alloc((size_t)N * D * 4);
    float* agg   = (float*)alloc((size_t)N * D * 4);   // reused as z2
    float* z1    = (float*)alloc((size_t)N * D2 * 4);
    bf16*  A1    = (bf16*)alloc((size_t)N * D * 2);
    bf16*  W1b   = (bf16*)alloc((size_t)L * D * D2 * 2);
    bf16*  W2b   = (bf16*)alloc((size_t)L * D2 * D * 2);
    float* stats = (float*)alloc(1536 * 4);
    float* sum1 = stats,        *sq1 = stats + 256;
    float* scale1 = stats + 512, *bias1 = stats + 768;
    float* sum2 = stats + 1024, *sq2 = stats + 1152;
    float* scale2 = stats + 1280, *bias2 = stats + 1408;
    float* z2 = agg;

    const int TB = 256;
    long nW = (long)L * D * D2;
    conv_bf16<<<dim3((unsigned)((nW + TB - 1) / TB)), TB, 0, stream>>>(W1, W1b, nW);
    conv_bf16<<<dim3((unsigned)((nW + TB - 1) / TB)), TB, 0, stream>>>(W2, W2b, nW);
    init_h<<<dim3((unsigned)((nND + TB - 1) / TB)), TB, 0, stream>>>(x, emb, h, nND);

    dim3 gridG1((N + 63) / 64, D2 / 64);
    dim3 gridG2((N + 63) / 64, D / 64);
    unsigned nodeBlocks = (unsigned)((nND + TB - 1) / TB);
    unsigned edgeBlocks = (unsigned)(((long)E * 32 + TB - 1) / TB);
    unsigned statBlocks = (unsigned)((N + 127) / 128);

    for (int l = 0; l < L; ++l) {
        zero_f32<<<nodeBlocks, TB, 0, stream>>>(agg, nND);
        zero_f32<<<6, TB, 0, stream>>>(stats, 1536);

        edge_gin<<<edgeBlocks, TB, 0, stream>>>(ei, ea, We + (long)l * 7 * D,
                                                be + (long)l * D, h, agg, E);
        node_pre<<<nodeBlocks, TB, 0, stream>>>(h, agg, eps, l, A1, nND);

        gemm_wmma<128, 256, false><<<gridG1, 128, 0, stream>>>(
            A1, nullptr, nullptr, nullptr,
            W1b + (long)l * D * D2, b1 + (long)l * D2, z1, N);

        col_stats<256><<<statBlocks, 256, 0, stream>>>(z1, sum1, sq1, N);
        bn_finalize<256><<<1, 256, 0, stream>>>(sum1, sq1, g1 + (long)l * D2,
                                                bt1 + (long)l * D2, scale1, bias1, N);

        gemm_wmma<256, 128, true><<<gridG2, 128, 0, stream>>>(
            nullptr, z1, scale1, bias1,
            W2b + (long)l * D2 * D, b2 + (long)l * D, z2, N);

        col_stats<128><<<statBlocks, 128, 0, stream>>>(z2, sum2, sq2, N);
        bn_finalize<128><<<1, 128, 0, stream>>>(sum2, sq2, gbn + (long)l * D,
                                                bbn + (long)l * D, scale2, bias2, N);

        float* dst = (l == L - 1) ? out : h;
        bn_apply<<<nodeBlocks, TB, 0, stream>>>(z2, scale2, bias2, dst, nND,
                                                (l < L - 1) ? 1 : 0);
    }
}
